// MKGNN_12575664242733
// MI455X (gfx1250) — matrix operand order repaired and checked by reading, not compiled, not run on hardware
//
#include <hip/hip_runtime.h>
#include <cstdint>

// ---------------------------------------------------------------------------
// MKGNN for MI455X (gfx1250, wave32, WMMA + async-to-LDS).
// Dominant work: two adj(4000x4000) @ X(4000x256) GEMMs -> bf16 WMMA with
// global_load_async_to_lds_b128 double-buffered staging.
// MHA / projection GEMMs -> f32->bf16 WMMA GEMM with fragment-ordered LDS.
// M=1 tails -> scalar memory-bound kernels.
// ---------------------------------------------------------------------------

typedef __attribute__((ext_vector_type(16))) __bf16 v16bf;
typedef __attribute__((ext_vector_type(4)))  __bf16 v4bf;
typedef __attribute__((ext_vector_type(8)))  float  v8f;

#define TBM 32
#define TBN 64
#define TBK 32

// A-fragment K permutation (ISA 7.12.2, 16-bit A 16x32):
// khalf0 slots 0..15 hold K{0..7,16..23}; khalf1 slots 16..31 hold K{8..15,24..31}
// => 8-wide K-block kb maps to slot-block perm(kb) = {0,2,1,3}[kb]
__device__ __forceinline__ int a_slot_of_k(int k) {
    return (k & 7) | ((k & 8) << 1) | ((k & 16) >> 1);
}

// ---------------------------------------------------------------------------
// Generic GEMM: C(M,N) = [relu?](A_f32(M,K) @ B_f32(K,N) + bias), opt relu(A).
// Requires M%32==0, K%32==0, N%64==0 (true at all call sites).
// ---------------------------------------------------------------------------
__global__ __launch_bounds__(256) void gemm_wmma_bf16(
    const float* __restrict__ A, const float* __restrict__ B,
    const float* __restrict__ bias, float* __restrict__ C,
    int M, int N, int K, int reluA, int reluOut)
{
    __shared__ __align__(64) __bf16 As[TBM][TBK];   // row-major (m,k), natural K order
    __shared__ __align__(64) __bf16 Bs[TBN][TBK];   // transposed (n-major)

    const int tid   = threadIdx.x;
    const int wave  = tid >> 5;
    const int lane  = tid & 31;
    const int bm    = blockIdx.x * TBM;
    const int bn    = blockIdx.y * TBN;
    const int wm    = (wave >> 2) << 4;
    const int wn    = (wave & 3)  << 4;
    const int l16   = lane & 15;
    const int khalf = lane >> 4;

    v8f acc = {0.f,0.f,0.f,0.f,0.f,0.f,0.f,0.f};

    for (int k0 = 0; k0 < K; k0 += TBK) {
        // A tile: 32x32 f32 = 256 float4, one per thread
        {
            int r  = tid >> 3;
            int c4 = (tid & 7) << 2;
            float4 v = *(const float4*)&A[(size_t)(bm + r)*K + k0 + c4];
            if (reluA) {
                v.x = fmaxf(v.x, 0.f); v.y = fmaxf(v.y, 0.f);
                v.z = fmaxf(v.z, 0.f); v.w = fmaxf(v.w, 0.f);
            }
            v4bf p; p[0]=(__bf16)v.x; p[1]=(__bf16)v.y; p[2]=(__bf16)v.z; p[3]=(__bf16)v.w;
            *(v4bf*)&As[r][c4] = p;   // 4 consecutive k (natural order)
        }
        // B tile: 32x64 f32 = 512 float4, two per thread; transpose into Bs[n][k]
        #pragma unroll
        for (int i = 0; i < 2; ++i) {
            int idx = tid + i*256;
            int r   = idx >> 4;
            int c4  = (idx & 15) << 2;
            float4 v = *(const float4*)&B[(size_t)(k0 + r)*N + bn + c4];
            Bs[c4+0][r] = (__bf16)v.x;
            Bs[c4+1][r] = (__bf16)v.y;
            Bs[c4+2][r] = (__bf16)v.z;
            Bs[c4+3][r] = (__bf16)v.w;
        }
        // gfx1250 global_prefetch of next K tile
        if (k0 + TBK < K) {
            __builtin_prefetch(&A[(size_t)(bm + (tid >> 3))*K + k0 + TBK], 0, 1);
            __builtin_prefetch(&B[(size_t)(k0 + TBK + (tid >> 3))*N + bn], 0, 1);
        }
        __syncthreads();

        v16bf a = *(const v16bf*)&As[wm + l16][khalf << 4];
        v16bf b = *(const v16bf*)&Bs[wn + l16][khalf << 4];
        acc = __builtin_amdgcn_wmma_f32_16x16x32_bf16(
                  false, a, false, b, (short)0, acc, false, false);
        __syncthreads();
    }

    const int mh = khalf * 8;
    #pragma unroll
    for (int i = 0; i < 8; ++i) {
        int gm = bm + wm + mh + i;
        int gn = bn + wn + l16;
        float v = acc[i];
        if (bias) v += bias[gn];
        if (reluOut) v = fmaxf(v, 0.0f);
        C[(size_t)gm*N + gn] = v;
    }
}

// ---------------------------------------------------------------------------
// Async GEMM: C(M,N) = [relu?](A_bf16(M,K) @ Bt_bf16(N,K)^T + bias)
// Tiles staged by global_load_async_to_lds_b128, double buffered.
// Requires M%32==0, K%32==0, N%64==0; all 16B-aligned rows (K%8==0).
// ---------------------------------------------------------------------------
__global__ __launch_bounds__(256) void gemm_wmma_async(
    const __bf16* __restrict__ A, const __bf16* __restrict__ Bt,
    const float* __restrict__ bias, float* __restrict__ C,
    int M, int N, int K, int reluOut)
{
    __shared__ __align__(64) __bf16 As[2][TBM][TBK];
    __shared__ __align__(64) __bf16 Bs[2][TBN][TBK];

    const int tid   = threadIdx.x;
    const int wave  = tid >> 5;
    const int lane  = tid & 31;
    const int bm    = blockIdx.x * TBM;
    const int bn    = blockIdx.y * TBN;
    const int wm    = (wave >> 2) << 4;
    const int wn    = (wave & 3)  << 4;
    const int l16   = lane & 15;
    const int khalf = lane >> 4;

    const int rr    = tid >> 2;                       // A row (tid<128) / Bt row
    const int kb    = tid & 3;                        // 8-wide K block
    const int aslot = kb << 3;  // natural k order

    auto issue = [&](int buf, int k0) {
        if (tid < 128) {   // A tile: 32 rows x 4 b128 = 128 transfers
            uint32_t lds = (uint32_t)(uintptr_t)&As[buf][rr][aslot];
            const __bf16* g = A + (size_t)(bm + rr)*K + k0 + kb*8;
            asm volatile("global_load_async_to_lds_b128 %0, %1, off"
                         :: "v"(lds), "v"(g) : "memory");
        }
        {                  // Bt tile: 64 rows x 4 b128 = 256 transfers
            uint32_t lds = (uint32_t)(uintptr_t)&Bs[buf][rr][kb << 3];
            const __bf16* g = Bt + (size_t)(bn + rr)*K + k0 + kb*8;
            asm volatile("global_load_async_to_lds_b128 %0, %1, off"
                         :: "v"(lds), "v"(g) : "memory");
        }
    };

    v8f acc = {0.f,0.f,0.f,0.f,0.f,0.f,0.f,0.f};
    const int NK = K / TBK;

    issue(0, 0);
    for (int it = 0; it < NK; ++it) {
        int buf = it & 1;
        asm volatile("s_wait_asynccnt 0x0" ::: "memory");
        __syncthreads();                       // tile ready; prev compute done
        if (it + 1 < NK) issue(buf ^ 1, (it + 1) * TBK);
        v16bf a = *(const v16bf*)&As[buf][wm + l16][khalf << 4];
        v16bf b = *(const v16bf*)&Bs[buf][wn + l16][khalf << 4];
        acc = __builtin_amdgcn_wmma_f32_16x16x32_bf16(
                  false, a, false, b, (short)0, acc, false, false);
    }

    const int mh = khalf * 8;
    #pragma unroll
    for (int i = 0; i < 8; ++i) {
        int gm = bm + wm + mh + i;
        int gn = bn + wn + l16;
        float v = acc[i];
        if (bias) v += bias[gn];
        if (reluOut) v = fmaxf(v, 0.0f);
        C[(size_t)gm*N + gn] = v;
    }
}

// f32 -> bf16 (n4 float4 groups)
__global__ void convert_bf16(const float* __restrict__ in, __bf16* __restrict__ out, int n4)
{
    int i = blockIdx.x * blockDim.x + threadIdx.x;
    if (i >= n4) return;
    float4 v = ((const float4*)in)[i];
    v4bf p; p[0]=(__bf16)v.x; p[1]=(__bf16)v.y; p[2]=(__bf16)v.z; p[3]=(__bf16)v.w;
    ((v4bf*)out)[i] = p;
}

// in (R,C) f32 row-major  ->  out (C,R) bf16 row-major
__global__ void transpose_convert(const float* __restrict__ in, __bf16* __restrict__ out,
                                  int R, int C)
{
    int i = blockIdx.x * blockDim.x + threadIdx.x;
    if (i >= R * C) return;
    int c = i / R, r = i % R;
    out[i] = (__bf16)in[(size_t)r*C + c];
}

// out[v,d] = 512*mean_l(emb[codes[v,l],d]) + pe0[d],  pe0 = 1 on odd d
__global__ __launch_bounds__(256) void gather_mean(
    const int* __restrict__ codes, const float* __restrict__ emb,
    float* __restrict__ out, int L)
{
    int v = blockIdx.x;
    for (int d = threadIdx.x; d < 512; d += 256) {
        float s = 0.f;
        for (int l = 0; l < L; ++l) s += emb[(size_t)codes[v*L + l]*512 + d];
        out[v*512 + d] = (512.0f * s) / (float)L + ((d & 1) ? 1.0f : 0.0f);
    }
}

__global__ void gather_prior(const int* __restrict__ codes,
                             const float* __restrict__ emb, float* __restrict__ out)
{
    for (int d = threadIdx.x; d < 512; d += 256) {
        float s = 0.f;
        for (int l = 0; l < 24; ++l) s += emb[(size_t)codes[l]*512 + d];
        out[d] = s / 24.0f;
    }
}

// One head of 64-token MHA; qkv row layout: [q(512)|k(512)|v(512)]
__global__ __launch_bounds__(256) void attention64(
    const float* __restrict__ qkv, float* __restrict__ out)
{
    const int h = blockIdx.x;
    __shared__ __bf16 ks[64][128];
    __shared__ __bf16 vs[64][128];
    __shared__ float  sc[64][64];
    int tid = threadIdx.x;

    for (int i = tid; i < 64*128; i += 256) {
        int r = i >> 7, d = i & 127;
        ks[r][d] = (__bf16)qkv[r*1536 + 512  + h*128 + d];
        vs[r][d] = (__bf16)qkv[r*1536 + 1024 + h*128 + d];
    }
    __syncthreads();
    for (int i = tid; i < 64*64; i += 256) {
        int qi = i >> 6, kj = i & 63;
        const float* qp = qkv + qi*1536 + h*128;
        float s = 0.f;
        for (int d = 0; d < 128; ++d) s += qp[d] * (float)ks[kj][d];
        sc[qi][kj] = s * 0.08838834764831845f;   // 1/sqrt(128)
    }
    __syncthreads();
    if (tid < 64) {
        float m = sc[tid][0];
        for (int j = 1; j < 64; ++j) m = fmaxf(m, sc[tid][j]);
        float ssum = 0.f;
        for (int j = 0; j < 64; ++j) { float e = __expf(sc[tid][j] - m); sc[tid][j] = e; ssum += e; }
        float inv = 1.0f / ssum;
        for (int j = 0; j < 64; ++j) sc[tid][j] *= inv;
    }
    __syncthreads();
    for (int i = tid; i < 64*128; i += 256) {
        int qi = i >> 7, d = i & 127;
        float s = 0.f;
        for (int kj = 0; kj < 64; ++kj) s += sc[qi][kj] * (float)vs[kj][d];
        out[qi*512 + h*128 + d] = s;
    }
}

// y[n] = [relu?](x) . W[:,n] + b[n],  W row-major (K,N)
__global__ void gemv_col(const float* __restrict__ x, const float* __restrict__ W,
                         const float* __restrict__ b, float* __restrict__ y,
                         int K, int N, int reluX)
{
    int n = blockIdx.x * blockDim.x + threadIdx.x;
    if (n >= N) return;
    float s = b ? b[n] : 0.0f;
    for (int k = 0; k < K; ++k) {
        float xv = x[k];
        if (reluX) xv = fmaxf(xv, 0.0f);
        s += xv * W[(size_t)k*N + n];
    }
    y[n] = s;
}

// y[r] = x . W[r,:]  (warp per row), W row-major (N,K)
__global__ void gemv_rowdot(const float* __restrict__ x, const float* __restrict__ W,
                            float* __restrict__ y, int K, int N)
{
    int warp = threadIdx.x >> 5;
    int lane = threadIdx.x & 31;
    int row  = blockIdx.x * 8 + warp;
    if (row >= N) return;
    const float* wr = W + (size_t)row*K;
    float s = 0.f;
    for (int k = lane; k < K; k += 32) s += x[k] * wr[k];
    for (int m = 16; m > 0; m >>= 1) s += __shfl_xor(s, m, 32);
    if (lane == 0) y[row] = s;
}

__global__ void softmax_inplace(float* __restrict__ x, int n)
{
    __shared__ float red[256];
    int tid = threadIdx.x;
    float m = -3.4e38f;
    for (int i = tid; i < n; i += 256) m = fmaxf(m, x[i]);
    red[tid] = m; __syncthreads();
    for (int s = 128; s > 0; s >>= 1) { if (tid < s) red[tid] = fmaxf(red[tid], red[tid+s]); __syncthreads(); }
    float mx = red[0]; __syncthreads();
    float acc = 0.f;
    for (int i = tid; i < n; i += 256) { float e = __expf(x[i] - mx); x[i] = e; acc += e; }
    red[tid] = acc; __syncthreads();
    for (int s = 128; s > 0; s >>= 1) { if (tid < s) red[tid] += red[tid+s]; __syncthreads(); }
    float inv = 1.0f / red[0]; __syncthreads();
    for (int i = tid; i < n; i += 256) x[i] *= inv;
}

__global__ void zero_f32(float* p, int n)
{
    int i = blockIdx.x * blockDim.x + threadIdx.x;
    if (i < n) p[i] = 0.f;
}

// fact2_voc[c] += vw[r] for each UNIQUE code c in med row r (one-hot .set semantics)
__global__ void hist_scatter(const float* __restrict__ vw, const int* __restrict__ med,
                             float* __restrict__ voc)
{
    int r = threadIdx.x;
    if (r >= 63) return;
    float w = vw[r];
    for (int j = 0; j < 24; ++j) {
        int c = med[r*24 + j];
        bool dup = false;
        for (int jj = 0; jj < j; ++jj) if (med[r*24 + jj] == c) { dup = true; break; }
        if (!dup) atomicAdd(&voc[c], w);
    }
}

__global__ void concat_pat(const float* __restrict__ o1, const float* __restrict__ o2,
                           float* __restrict__ pat)
{
    int i = blockIdx.x * blockDim.x + threadIdx.x;
    if (i >= 64*1024) return;
    int v = i >> 10, d = i & 1023;
    pat[i] = (d < 512) ? o1[v*512 + d] : o2[v*512 + (d - 512)];
}

__global__ void build_cat(const float* __restrict__ o1, const float* __restrict__ o2,
                          const float* __restrict__ pk, const float* __restrict__ f1,
                          const float* __restrict__ f2, float* __restrict__ cat)
{
    int i = blockIdx.x * blockDim.x + threadIdx.x;
    if (i >= 2048) return;
    if      (i < 512)  cat[i] = o1[63*512 + i];
    else if (i < 1024) cat[i] = o2[63*512 + (i - 512)];
    else if (i < 1536) cat[i] = pk[i - 1024];
    else if (i < 1792) cat[i] = f1[i - 1536];
    else               cat[i] = f2[i - 1792];
}

extern "C" void kernel_launch(void* const* d_in, const int* in_sizes, int n_in,
                              void* d_out, int out_size, void* d_ws, size_t ws_size,
                              hipStream_t stream)
{
    (void)in_sizes; (void)n_in; (void)out_size; (void)ws_size;
    const int*   diag  = (const int*)d_in[0];
    const int*   proc  = (const int*)d_in[1];
    const int*   med   = (const int*)d_in[2];
    const int*   prior = (const int*)d_in[3];
    const float* emb0  = (const float*)d_in[4];
    const float* emb1  = (const float*)d_in[5];
    const float* emb2  = (const float*)d_in[6];
    const float* wi    = (const float*)d_in[7];   // (3,512,1536)
    const float* bi    = (const float*)d_in[8];   // (3,1536)
    const float* wo    = (const float*)d_in[9];   // (3,512,512)
    const float* bo    = (const float*)d_in[10];  // (3,512)
    const float* q1w   = (const float*)d_in[11];  // (1024,256)
    const float* q1b   = (const float*)d_in[12];
    const float* adj   = (const float*)d_in[13];  // (4000,4000)
    const float* g1w   = (const float*)d_in[14];  // (4000,256)
    const float* g1b   = (const float*)d_in[15];
    const float* g2w   = (const float*)d_in[16];  // (256,256)
    const float* g2b   = (const float*)d_in[17];
    const float* ow1   = (const float*)d_in[18];  // (2048,512)
    const float* ob1   = (const float*)d_in[19];
    const float* ow2   = (const float*)d_in[20];  // (512,4000)
    const float* ob2   = (const float*)d_in[21];
    float* out = (float*)d_out;

    float* ws = (float*)d_ws;
    float* i1   = ws;              // 64*512
    float* i2   = i1   + 64*512;
    float* pri  = i2   + 64*512;   // 512
    float* qkv1 = pri  + 512;      // 64*1536
    float* qkv2 = qkv1 + 64*1536;
    float* qkv3 = qkv2 + 64*1536;  // 1536
    float* att1 = qkv3 + 1536;     // 64*512
    float* att2 = att1 + 64*512;
    float* o1   = att2 + 64*512;
    float* o2   = o1   + 64*512;
    float* pat  = o2   + 64*512;   // 64*1024
    float* q1   = pat  + 64*1024;  // 64*256  (queries1)
    float* h    = q1   + 64*256;   // 4000*256
    float* hb   = h    + 4000*256;
    float* dm   = hb   + 4000*256;
    float* kw   = dm   + 4000*256; // 4000
    float* vw   = kw   + 4000;     // 64
    float* f2v  = vw   + 64;       // 4000
    float* f1   = f2v  + 4000;     // 256
    float* f2   = f1   + 256;      // 256
    float* cat  = f2   + 256;      // 2048
    float* o1h  = cat  + 2048;     // 512 (hidden out1)
    float* pk   = o1h  + 512;      // 512 (prior_knowledge)
    __bf16* adj_bf = (__bf16*)(pk + 512);        // 16,000,000 bf16 (16B aligned)
    __bf16* g1wT   = adj_bf + 16000000;          // (256,4000) bf16
    __bf16* hbT    = g1wT   + 1024000;           // (256,4000) bf16

    // bf16 pre-conversions for the async GCN GEMMs
    convert_bf16<<<(4000000 + 255)/256, 256, 0, stream>>>(adj, adj_bf, 4000000);
    transpose_convert<<<(1024000 + 255)/256, 256, 0, stream>>>(g1w, g1wT, 4000, 256);

    // Embedding gathers
    gather_mean <<<64, 256, 0, stream>>>(diag, emb0, i1, 48);
    gather_mean <<<64, 256, 0, stream>>>(proc, emb1, i2, 32);
    gather_prior<<<1, 256, 0, stream>>>(prior, emb2, pri);

    // MHA input projections (WMMA)
    gemm_wmma_bf16<<<dim3(2, 24), 256, 0, stream>>>(i1, wi,              bi,        qkv1, 64, 1536, 512, 0, 0);
    gemm_wmma_bf16<<<dim3(2, 24), 256, 0, stream>>>(i2, wi + 512*1536,   bi + 1536, qkv2, 64, 1536, 512, 0, 0);
    gemv_col<<<6, 256, 0, stream>>>(pri, wi + 2*512*1536, bi + 2*1536, qkv3, 512, 1536, 0);

    // Attention (one block per head)
    attention64<<<4, 256, 0, stream>>>(qkv1, att1);
    attention64<<<4, 256, 0, stream>>>(qkv2, att2);

    // MHA output projections (WMMA); stream2 with S=1 => o = v @ wo + bo
    gemm_wmma_bf16<<<dim3(2, 8), 256, 0, stream>>>(att1, wo,             bo,       o1, 64, 512, 512, 0, 0);
    gemm_wmma_bf16<<<dim3(2, 8), 256, 0, stream>>>(att2, wo + 512*512,   bo + 512, o2, 64, 512, 512, 0, 0);
    gemv_col<<<2, 256, 0, stream>>>(qkv3 + 1024, wo + 2*512*512, bo + 2*512, pk, 512, 512, 0);

    // queries1 = relu(concat(o1,o2)) @ q1_w + q1_b
    concat_pat<<<256, 256, 0, stream>>>(o1, o2, pat);
    gemm_wmma_bf16<<<dim3(2, 4), 256, 0, stream>>>(pat, q1w, q1b, q1, 64, 256, 1024, 1, 0);

    // GCN over drug vocab: dominant FLOPs, async-to-LDS + WMMA
    gemm_wmma_async<<<dim3(125, 4), 256, 0, stream>>>(adj_bf, g1wT, g1b, h, 4000, 256, 4000, 1);
    gemm_wmma_bf16 <<<dim3(125, 4), 256, 0, stream>>>(h, g2w, nullptr, hb, 4000, 256, 256, 0, 0);
    transpose_convert<<<(1024000 + 255)/256, 256, 0, stream>>>(hb, hbT, 4000, 256);
    gemm_wmma_async<<<dim3(125, 4), 256, 0, stream>>>(adj_bf, hbT, g2b, dm, 4000, 256, 4000, 0);

    // fact1 = softmax(query1 @ dm^T) @ dm
    gemv_rowdot<<<500, 256, 0, stream>>>(q1 + 63*256, dm, kw, 256, 4000);
    softmax_inplace<<<1, 256, 0, stream>>>(kw, 4000);
    gemv_col<<<1, 256, 0, stream>>>(kw, dm, nullptr, f1, 4000, 256, 0);

    // fact2 = softmax(query1 @ hist_keys^T) @ hist_values @ dm
    gemv_rowdot<<<8, 256, 0, stream>>>(q1 + 63*256, q1, vw, 256, 63);
    softmax_inplace<<<1, 256, 0, stream>>>(vw, 63);
    zero_f32<<<16, 256, 0, stream>>>(f2v, 4000);
    hist_scatter<<<1, 64, 0, stream>>>(vw, med, f2v);
    gemv_col<<<1, 256, 0, stream>>>(f2v, dm, nullptr, f2, 4000, 256, 0);

    // Output MLP
    build_cat<<<8, 256, 0, stream>>>(o1, o2, pk, f1, f2, cat);
    gemv_col<<<2, 256, 0, stream>>>(cat, ow1, ob1, o1h, 2048, 512, 1);
    gemv_col<<<16, 256, 0, stream>>>(o1h, ow2, ob2, out, 512, 4000, 1);
}
